// Nested_k321_GNN_56616258896131
// MI455X (gfx1250) — matrix operand order repaired and checked by reading, not compiled, not run on hardware
//
#include <hip/hip_runtime.h>
#include <math.h>

typedef __attribute__((ext_vector_type(16))) _Float16 v16h;
typedef __attribute__((ext_vector_type(8)))  _Float16 v8h;
typedef __attribute__((ext_vector_type(8)))  float    v8f;

#define GF_ACCUM 1
#define GF_BIAS  2
#define GF_ELU   4

#define S_   20000L
#define G_   256
#define N_   200000L
#define N2_  300000L
#define N3_  400000L
#define Ecnt  600000L
#define E2cnt 1200000L
#define E3cnt 1600000L
#define R_   5

#define MAX_KP 192   // max (padded) K across all GEMMs

__device__ __forceinline__ float elu1(float v) { return v > 0.f ? v : (__expf(v) - 1.f); }

// ---------------------------------------------------------------------------
// WMMA GEMM: C[M x (NT*16)] = op( A[M x Kp] @ B[Kp x (NT*16)] ).
// NT is compile-time (no dynamic acc indexing -> no spills).
// Requirements: Kp % 32 == 0, lda % 4 == 0 (float4 loads).
// B staged in LDS transposed [n][k] as f16 so each lane fragment is two 16-byte
// runs at +0/+32 bytes (ds_load_b128 x2). One wave computes a 16-row strip across
// all NT N-tiles, sharing one A fragment per K-step.
// ---------------------------------------------------------------------------
template <int NT>
__global__ void wmma_gemm_t(const float* __restrict__ A, int lda,
                            const float* __restrict__ B, int ldb,
                            float* __restrict__ C, int ldc,
                            const float* __restrict__ bias,
                            int M, int Kp, int flags, int nTilesM)
{
    __shared__ _Float16 Bsh[MAX_KP * 16 * NT];
    const int Ncols = NT * 16;

    // cooperative B load: f32 -> f16, transposed to [n][k]
    for (int idx = threadIdx.x; idx < Kp * Ncols; idx += blockDim.x) {
        int n = idx / Kp, k = idx % Kp;
        Bsh[idx] = (_Float16)B[(long)k * ldb + n];
    }
    __syncthreads();

    int mTile = blockIdx.x * (blockDim.x >> 5) + (threadIdx.x >> 5);
    if (mTile >= nTilesM) return;

    const int lane = threadIdx.x & 31;
    const int half = lane >> 4;
    const int lmod = lane & 15;
    int m = mTile * 16 + lmod;
    if (m >= M) m = M - 1;                      // clamp tail rows; stores guarded below
    const float* __restrict__ Arow = A + (long)m * lda;

    // per-lane B fragment base pointers (K-invariant)
    const _Float16* bp[NT];
#pragma unroll
    for (int nt = 0; nt < NT; ++nt)
        bp[nt] = &Bsh[(nt * 16 + lmod) * Kp + half * 8];

    v8f acc[NT];
#pragma unroll
    for (int nt = 0; nt < NT; ++nt) acc[nt] = v8f{0.f,0.f,0.f,0.f,0.f,0.f,0.f,0.f};

    for (int k0 = 0; k0 < Kp; k0 += 32) {
        // A fragment: elems 0..7 -> K = k0 + 8*half + j ; elems 8..15 -> +16
        const float4* ap0 = (const float4*)(Arow + k0 + half * 8);
        const float4* ap1 = (const float4*)(Arow + k0 + 16 + half * 8);
        float4 a0 = ap0[0], a1 = ap0[1];
        float4 a2 = ap1[0], a3 = ap1[1];
        v16h af;
        af[0]  = (_Float16)a0.x; af[1]  = (_Float16)a0.y; af[2]  = (_Float16)a0.z; af[3]  = (_Float16)a0.w;
        af[4]  = (_Float16)a1.x; af[5]  = (_Float16)a1.y; af[6]  = (_Float16)a1.z; af[7]  = (_Float16)a1.w;
        af[8]  = (_Float16)a2.x; af[9]  = (_Float16)a2.y; af[10] = (_Float16)a2.z; af[11] = (_Float16)a2.w;
        af[12] = (_Float16)a3.x; af[13] = (_Float16)a3.y; af[14] = (_Float16)a3.z; af[15] = (_Float16)a3.w;

#pragma unroll
        for (int nt = 0; nt < NT; ++nt) {
            const _Float16* p = bp[nt] + k0;
            v8h b0 = *(const v8h*)p;
            v8h b1 = *(const v8h*)(p + 16);
            v16h bf;
#pragma unroll
            for (int j = 0; j < 8; ++j) { bf[j] = b0[j]; bf[j + 8] = b1[j]; }
            acc[nt] = __builtin_amdgcn_wmma_f32_16x16x32_f16(
                false, af, false, bf, (short)0, acc[nt], false, false);
        }
    }

#pragma unroll
    for (int nt = 0; nt < NT; ++nt) {
        int n = nt * 16 + lmod;
        float bv = (flags & GF_BIAS) ? bias[n] : 0.f;
#pragma unroll
        for (int r = 0; r < 8; ++r) {
            int mr = mTile * 16 + r + half * 8;   // C/D layout: VGPR r -> row r (+8 hi half)
            if (mr < M) {
                long off = (long)mr * ldc + n;
                float v = acc[nt][r];
                if (flags & GF_ACCUM) v += C[off];
                v += bv;
                if (flags & GF_ELU) v = elu1(v);
                C[off] = v;
            }
        }
    }
}

// ---------------------------------------------------------------------------
// Elementwise / gather / scatter kernels
// ---------------------------------------------------------------------------
__global__ void fill_f32(float* p, float v, long n) {
    long i = (long)blockIdx.x * blockDim.x + threadIdx.x;
    if (i < n) p[i] = v;
}

// dst[0:kn_total) = src zero-padded past kn_valid (rows contiguous)
__global__ void pad_weight(float* __restrict__ dst, const float* __restrict__ src,
                           int kn_valid, int kn_total) {
    int t = blockIdx.x * blockDim.x + threadIdx.x;
    if (t >= kn_total) return;
    dst[t] = (t < kn_valid) ? src[t] : 0.f;
}

// Y[i,:] (=|+=) T[rowOff + idx[i], :]
__global__ void rows_from_table(float* __restrict__ Y, const int* __restrict__ idx,
                                const float* __restrict__ T, int C, int rowOff,
                                long M, int accum) {
    long t = (long)blockIdx.x * blockDim.x + threadIdx.x;
    if (t >= M * C) return;
    long i = t / C; int c = (int)(t % C);
    float v = T[(long)(rowOff + idx[i]) * C + c];
    if (accum) Y[t] += v; else Y[t] = v;
}

// Y[dst[e],:] += T[rowOff + iso[src[e]], :]   (one-hot @ Wn aggregation)
__global__ void edge_table_add(float* __restrict__ Y, const int* __restrict__ src,
                               const int* __restrict__ dst, const int* __restrict__ iso,
                               const float* __restrict__ T, int C, int rowOff, long E) {
    long t = (long)blockIdx.x * blockDim.x + threadIdx.x;
    if (t >= E * C) return;
    long e = t / C; int c = (int)(t % C);
    float v = T[(long)(rowOff + iso[src[e]]) * C + c];
    atomicAdd(&Y[(long)dst[e] * C + c], v);
}

// agg[dst[e],:] += X[src[e],:]
__global__ void edge_feat_add(float* __restrict__ agg, const int* __restrict__ src,
                              const int* __restrict__ dst, const float* __restrict__ X,
                              int C, long E) {
    long t = (long)blockIdx.x * blockDim.x + threadIdx.x;
    if (t >= E * C) return;
    long e = t / C; int c = (int)(t % C);
    atomicAdd(&agg[(long)dst[e] * C + c], X[(long)src[e] * C + c]);
}

// hn[dst[e],:] += ea[e,r] * Xr[src[e],:]
__global__ void edge_rel_add(float* __restrict__ hn, const int* __restrict__ src,
                             const int* __restrict__ dst, const float* __restrict__ ea,
                             int Rr, int r, const float* __restrict__ Xr, int C, long E) {
    long t = (long)blockIdx.x * blockDim.x + threadIdx.x;
    if (t >= E * C) return;
    long e = t / C; int c = (int)(t % C);
    atomicAdd(&hn[(long)dst[e] * C + c], ea[e * Rr + r] * Xr[(long)src[e] * C + c]);
}

__global__ void bias_act(float* __restrict__ Y, const float* __restrict__ bias,
                         int C, long M, int doElu) {
    long t = (long)blockIdx.x * blockDim.x + threadIdx.x;
    if (t >= M * C) return;
    int c = (int)(t % C);
    float v = Y[t] + (bias ? bias[c] : 0.f);
    if (doElu) v = elu1(v);
    Y[t] = v;
}

// out[seg[i],:] += v[i,:] ; cnt[seg[i]] += 1
__global__ void seg_sum(float* __restrict__ out, float* __restrict__ cnt,
                        const int* __restrict__ seg, const float* __restrict__ v,
                        int C, long M) {
    long t = (long)blockIdx.x * blockDim.x + threadIdx.x;
    if (t >= M * C) return;
    long i = t / C; int c = (int)(t % C);
    int s = seg[i];
    atomicAdd(&out[(long)s * C + c], v[i * (long)C + c]);
    if (c == 0) atomicAdd(&cnt[s], 1.f);
}

// out[segIdx[j],:] += v[srcIdx[j],:] ; cnt[segIdx[j]] += 1   (avg_pool reverse)
__global__ void seg_sum_gather(float* __restrict__ out, float* __restrict__ cnt,
                               const int* __restrict__ segIdx, const int* __restrict__ srcIdx,
                               const float* __restrict__ v, int C, long M) {
    long t = (long)blockIdx.x * blockDim.x + threadIdx.x;
    if (t >= M * C) return;
    long j = t / C; int c = (int)(t % C);
    int s = segIdx[j];
    atomicAdd(&out[(long)s * C + c], v[(long)srcIdx[j] * C + c]);
    if (c == 0) atomicAdd(&cnt[s], 1.f);
}

__global__ void seg_div(float* __restrict__ out, const float* __restrict__ cnt,
                        int C, long Nseg) {
    long t = (long)blockIdx.x * blockDim.x + threadIdx.x;
    if (t >= Nseg * C) return;
    long s = t / C;
    out[t] /= fmaxf(cnt[s], 1.f);
}

// Xcat[i, 0:8]=z_table[z[i]], [8:21]=x[i], [21:85]=hNP[i], [85:96]=0  (lda=96)
__global__ void concat_node(float* __restrict__ Xcat, const float* __restrict__ zt,
                            const int* __restrict__ z, const float* __restrict__ x,
                            const float* __restrict__ hNP, long M) {
    long t = (long)blockIdx.x * blockDim.x + threadIdx.x;
    if (t >= M * 96) return;
    long i = t / 96; int c = (int)(t % 96);
    float v;
    if (c < 8)       v = zt[(long)z[i] * 8 + c];
    else if (c < 21) v = x[i * 13 + (c - 8)];
    else if (c < 85) v = hNP[i * 64 + (c - 21)];
    else             v = 0.f;
    Xcat[t] = v;
}

// xg[s2g[s], 0:192] += [x1|x2|x3][s,:] ; cnt[g] += 1
__global__ void concat_seg_sum(float* __restrict__ xg, float* __restrict__ cnt,
                               const int* __restrict__ s2g,
                               const float* __restrict__ x1, const float* __restrict__ x2,
                               const float* __restrict__ x3, long Ssz) {
    long t = (long)blockIdx.x * blockDim.x + threadIdx.x;
    if (t >= Ssz * 192) return;
    long s = t / 192; int c = (int)(t % 192);
    float v = (c < 64) ? x1[s * 64 + c]
            : (c < 128) ? x2[s * 64 + (c - 64)]
                        : x3[s * 64 + (c - 128)];
    int g = s2g[s];
    atomicAdd(&xg[(long)g * 192 + c], v);
    if (c == 0) atomicAdd(&cnt[g], 1.f);
}

__global__ void fc3_kernel(const float* __restrict__ h, const float* __restrict__ W,
                           const float* __restrict__ b, float* __restrict__ out) {
    int g = blockIdx.x * blockDim.x + threadIdx.x;
    if (g >= G_) return;
    float acc = b[0];
#pragma unroll
    for (int c = 0; c < 32; ++c) acc += h[g * 32 + c] * W[c];
    out[g] = acc;
}

// ---------------------------------------------------------------------------
static inline unsigned cdivu(long a, long b) { return (unsigned)((a + b - 1) / b); }

extern "C" void kernel_launch(void* const* d_in, const int* in_sizes, int n_in,
                              void* d_out, int out_size, void* d_ws, size_t ws_size,
                              hipStream_t stream) {
    // ---- inputs (setup_inputs order) ----
    const float* x    = (const float*)d_in[0];
    const int*   z    = (const int*)d_in[1];
    const int*   ei   = (const int*)d_in[2];     // (2,E)
    const float* ea   = (const float*)d_in[3];   // (E,5)
    const int*   iso2 = (const int*)d_in[4];
    const int*   iso3 = (const int*)d_in[5];
    const int*   ei2  = (const int*)d_in[6];     // (2,E2)
    const int*   ei3  = (const int*)d_in[7];     // (2,E3)
    const int*   ai2  = (const int*)d_in[8];     // (2,600000)
    const int*   ai3  = (const int*)d_in[9];     // (2,1200000)
    const int*   a2s  = (const int*)d_in[10];
    const int*   a3s  = (const int*)d_in[11];
    const int*   n2s  = (const int*)d_in[12];
    const int*   s2g  = (const int*)d_in[13];
    const float* zt   = (const float*)d_in[14];
    const float* c1W = (const float*)d_in[15]; const float* c1R = (const float*)d_in[16]; const float* c1b = (const float*)d_in[17];
    const float* c2W = (const float*)d_in[18]; const float* c2R = (const float*)d_in[19]; const float* c2b = (const float*)d_in[20];
    const float* c3W = (const float*)d_in[21]; const float* c3R = (const float*)d_in[22]; const float* c3b = (const float*)d_in[23];
    const float* c4Wn = (const float*)d_in[24]; const float* c4Wr = (const float*)d_in[25]; const float* c4b = (const float*)d_in[26];
    const float* c5Wn = (const float*)d_in[27]; const float* c5Wr = (const float*)d_in[28]; const float* c5b = (const float*)d_in[29];
    const float* c6Wn = (const float*)d_in[30]; const float* c6Wr = (const float*)d_in[31]; const float* c6b = (const float*)d_in[32];
    const float* c7Wn = (const float*)d_in[33]; const float* c7Wr = (const float*)d_in[34]; const float* c7b = (const float*)d_in[35];
    const float* f1W = (const float*)d_in[36]; const float* f1b = (const float*)d_in[37];
    const float* f2W = (const float*)d_in[38]; const float* f2b = (const float*)d_in[39];
    const float* f3W = (const float*)d_in[40]; const float* f3b = (const float*)d_in[41];
    float* out = (float*)d_out;

    const int* e_src  = ei;        const int* e_dst  = ei  + Ecnt;
    const int* e2_src = ei2;       const int* e2_dst = ei2 + E2cnt;
    const int* e3_src = ei3;       const int* e3_dst = ei3 + E3cnt;
    const int* ai2_seg = ai2;      const int* ai2_src = ai2 + 600000;
    const int* ai3_seg = ai3;      const int* ai3_src = ai3 + 1200000;

    // ---- workspace arenas (floats); lifetimes phase-disjoint ----
    float* ws = (float*)d_ws;
    long o = 0;
    float* A0 = ws + o; o += 25600000;   // h3A ; h2A ; Xr (per-relation)
    float* A1 = ws + o; o += 25600000;   // h3B ; h2B ; hn1(+6.4M: hn2)
    float* A2 = ws + o; o += 32000000;   // agg3 ; agg2 ; Xcat(N x 96) ; hn3
    float* P0 = ws + o; o += 19200000;   // h2P ; hNP
    float* x1b = ws + o; o += S_ * 64;
    float* x2b = ws + o; o += S_ * 64;
    float* x3b = ws + o; o += S_ * 64;
    float* xg  = ws + o; o += (long)G_ * 192;
    float* fcA = ws + o; o += (long)G_ * 64;
    float* fcB = ws + o; o += (long)G_ * 32;
    float* Bpad = ws + o; o += 96 * 32;  // zero-padded conv1 weight staging
    float* cnt = ws + o; o += 400000;
    (void)ws_size; (void)n_in; (void)in_sizes; (void)out_size;

    const unsigned TB = 256;
    auto gemm = [&](const float* A, int lda, const float* B, int ldb, float* C, int ldc,
                    const float* bias, long M, int Kp, int Ncols, int flags) {
        int nTilesM = (int)((M + 15) / 16);
        unsigned blocks = cdivu(nTilesM, 8);
        if (Ncols == 32)
            wmma_gemm_t<2><<<blocks, TB, 0, stream>>>(A, lda, B, ldb, C, ldc, bias,
                                                      (int)M, Kp, flags, nTilesM);
        else
            wmma_gemm_t<4><<<blocks, TB, 0, stream>>>(A, lda, B, ldb, C, ldc, bias,
                                                      (int)M, Kp, flags, nTilesM);
    };
    auto fill = [&](float* p, long n) { fill_f32<<<cdivu(n, TB), TB, 0, stream>>>(p, 0.f, n); };

    // ================= Level-3 tuples =================
    // conv6 on one-hot input: h3A = elu( Wr6[iso3] + edge-scatter(Wn6[iso3[src]]) + b6 )
    float* h3A = A0;
    rows_from_table<<<cdivu(N3_ * 64, TB), TB, 0, stream>>>(h3A, iso3, c6Wr, 64, 0, N3_, 0);
    edge_table_add<<<cdivu(E3cnt * 64, TB), TB, 0, stream>>>(h3A, e3_src, e3_dst, iso3, c6Wn, 64, 0, E3cnt);
    bias_act<<<cdivu(N3_ * 64, TB), TB, 0, stream>>>(h3A, c6b, 64, N3_, 1);

    // conv7: h3B = elu( agg(h3A)@Wn7 + h3A@Wr7 + b7 )
    float* agg3 = A2;  float* h3B = A1;
    fill(agg3, N3_ * 64);
    edge_feat_add<<<cdivu(E3cnt * 64, TB), TB, 0, stream>>>(agg3, e3_src, e3_dst, h3A, 64, E3cnt);
    gemm(agg3, 64, c7Wn, 64, h3B, 64, nullptr, N3_, 64, 64, 0);
    gemm(h3A, 64, c7Wr, 64, h3B, 64, c7b, N3_, 64, 64, GF_ACCUM | GF_BIAS | GF_ELU);

    // x_3 = seg_mean(h3B, a3s, S)
    fill(x3b, S_ * 64); fill(cnt, S_);
    seg_sum<<<cdivu(N3_ * 64, TB), TB, 0, stream>>>(x3b, cnt, a3s, h3B, 64, N3_);
    seg_div<<<cdivu(S_ * 64, TB), TB, 0, stream>>>(x3b, cnt, 64, S_);

    // pool3: h2P[N2] = seg_mean(h3B[ai3_src], ai3_seg, N2)
    float* h2P = P0;
    fill(h2P, N2_ * 64); fill(cnt, N2_);
    seg_sum_gather<<<cdivu(1200000L * 64, TB), TB, 0, stream>>>(h2P, cnt, ai3_seg, ai3_src, h3B, 64, 1200000L);
    seg_div<<<cdivu(N2_ * 64, TB), TB, 0, stream>>>(h2P, cnt, 64, N2_);

    // ================= Level-2 tuples =================
    // conv4 on concat([h2P, onehot(iso2)]): 114x64 weights split [0:64] dense + [64:114] one-hot rows
    float* agg2 = A2;  float* h2A = A0;
    fill(agg2, N2_ * 64);
    edge_feat_add<<<cdivu(E2cnt * 64, TB), TB, 0, stream>>>(agg2, e2_src, e2_dst, h2P, 64, E2cnt);
    gemm(agg2, 64, c4Wn, 64, h2A, 64, nullptr, N2_, 64, 64, 0);                          // agg_h @ Wn[:64]
    edge_table_add<<<cdivu(E2cnt * 64, TB), TB, 0, stream>>>(h2A, e2_src, e2_dst, iso2, c4Wn, 64, 64, E2cnt);
    gemm(h2P, 64, c4Wr, 64, h2A, 64, nullptr, N2_, 64, 64, GF_ACCUM);                    // h @ Wr[:64]
    rows_from_table<<<cdivu(N2_ * 64, TB), TB, 0, stream>>>(h2A, iso2, c4Wr, 64, 64, N2_, 1);
    bias_act<<<cdivu(N2_ * 64, TB), TB, 0, stream>>>(h2A, c4b, 64, N2_, 1);

    // conv5
    float* h2B = A1;
    fill(agg2, N2_ * 64);
    edge_feat_add<<<cdivu(E2cnt * 64, TB), TB, 0, stream>>>(agg2, e2_src, e2_dst, h2A, 64, E2cnt);
    gemm(agg2, 64, c5Wn, 64, h2B, 64, nullptr, N2_, 64, 64, 0);
    gemm(h2A, 64, c5Wr, 64, h2B, 64, c5b, N2_, 64, 64, GF_ACCUM | GF_BIAS | GF_ELU);

    // x_2 = seg_mean(h2B, a2s, S)
    fill(x2b, S_ * 64); fill(cnt, S_);
    seg_sum<<<cdivu(N2_ * 64, TB), TB, 0, stream>>>(x2b, cnt, a2s, h2B, 64, N2_);
    seg_div<<<cdivu(S_ * 64, TB), TB, 0, stream>>>(x2b, cnt, 64, S_);

    // pool2: hNP[N] = seg_mean(h2B[ai2_src], ai2_seg, N)
    float* hNP = P0;
    fill(hNP, N_ * 64); fill(cnt, N_);
    seg_sum_gather<<<cdivu(600000L * 64, TB), TB, 0, stream>>>(hNP, cnt, ai2_seg, ai2_src, h2B, 64, 600000L);
    seg_div<<<cdivu(N_ * 64, TB), TB, 0, stream>>>(hNP, cnt, 64, N_);

    // ================= Node level (RGCN x3) =================
    float* Xcat = A2;   // N x 96 (cols 85..95 zero)
    concat_node<<<cdivu(N_ * 96, TB), TB, 0, stream>>>(Xcat, zt, z, x, hNP, N_);

    float* hn1 = A1;                 // N x 32
    float* hn2 = A1 + 6400000;       // N x 64
    float* Xr  = A0;                 // per-relation N x Nc

    // rgcn1: K=85 (padded to 96) -> 32
    pad_weight<<<cdivu(96 * 32, TB), TB, 0, stream>>>(Bpad, c1R, 85 * 32, 96 * 32);
    gemm(Xcat, 96, Bpad, 32, hn1, 32, nullptr, N_, 96, 32, 0);
    for (int r = 0; r < R_; ++r) {
        pad_weight<<<cdivu(96 * 32, TB), TB, 0, stream>>>(Bpad, c1W + (long)r * 85 * 32, 85 * 32, 96 * 32);
        gemm(Xcat, 96, Bpad, 32, Xr, 32, nullptr, N_, 96, 32, 0);
        edge_rel_add<<<cdivu(Ecnt * 32, TB), TB, 0, stream>>>(hn1, e_src, e_dst, ea, R_, r, Xr, 32, Ecnt);
    }
    bias_act<<<cdivu(N_ * 32, TB), TB, 0, stream>>>(hn1, c1b, 32, N_, 1);

    // rgcn2: 32 -> 64
    gemm(hn1, 32, c2R, 64, hn2, 64, nullptr, N_, 32, 64, 0);
    for (int r = 0; r < R_; ++r) {
        gemm(hn1, 32, c2W + (long)r * 32 * 64, 64, Xr, 64, nullptr, N_, 32, 64, 0);
        edge_rel_add<<<cdivu(Ecnt * 64, TB), TB, 0, stream>>>(hn2, e_src, e_dst, ea, R_, r, Xr, 64, Ecnt);
    }
    bias_act<<<cdivu(N_ * 64, TB), TB, 0, stream>>>(hn2, c2b, 64, N_, 1);

    // rgcn3: 64 -> 64  (Xcat dead, reuse A2 for hn3)
    float* hn3 = A2;
    gemm(hn2, 64, c3R, 64, hn3, 64, nullptr, N_, 64, 64, 0);
    for (int r = 0; r < R_; ++r) {
        gemm(hn2, 64, c3W + (long)r * 64 * 64, 64, Xr, 64, nullptr, N_, 64, 64, 0);
        edge_rel_add<<<cdivu(Ecnt * 64, TB), TB, 0, stream>>>(hn3, e_src, e_dst, ea, R_, r, Xr, 64, Ecnt);
    }
    bias_act<<<cdivu(N_ * 64, TB), TB, 0, stream>>>(hn3, c3b, 64, N_, 1);

    // x_1 = seg_mean(hn3, node_to_subgraph, S)
    fill(x1b, S_ * 64); fill(cnt, S_);
    seg_sum<<<cdivu(N_ * 64, TB), TB, 0, stream>>>(x1b, cnt, n2s, hn3, 64, N_);
    seg_div<<<cdivu(S_ * 64, TB), TB, 0, stream>>>(x1b, cnt, 64, S_);

    // ================= Readout =================
    fill(xg, (long)G_ * 192); fill(cnt, G_);
    concat_seg_sum<<<cdivu(S_ * 192, TB), TB, 0, stream>>>(xg, cnt, s2g, x1b, x2b, x3b, S_);
    seg_div<<<cdivu((long)G_ * 192, TB), TB, 0, stream>>>(xg, cnt, 192, G_);

    gemm(xg, 192, f1W, 64, fcA, 64, f1b, G_, 192, 64, GF_BIAS | GF_ELU);
    gemm(fcA, 64, f2W, 32, fcB, 32, f2b, G_, 64, 32, GF_BIAS | GF_ELU);
    fc3_kernel<<<1, 256, 0, stream>>>(fcB, f3W, f3b, out);
}
